// Temporal_Group_Residual_Module_60464549593758
// MI455X (gfx1250) — matrix operand order repaired and checked by reading, not compiled, not run on hardware
//
#include <hip/hip_runtime.h>
#include <hip/hip_bf16.h>
#include <stdint.h>

// ---------------------------------------------------------------------------
// Temporal Group Residual Module on MI455X (gfx1250, wave32, WMMA).
// GEMMs in bf16 via v_wmma_f32_16x16x32_bf16 (f32 accum). h kept bf16 in a
// WMMA A-fragment swizzled layout; LayerNorm+ReLU applied in place on it.
// B operand staged through double-buffered LDS (shared by all 8 waves of a
// block) to kill 8x redundant L2 traffic. Workspace ~118 MB.
// ---------------------------------------------------------------------------

#define M_TOTAL 12288   // 8*12*128 rows
#define HID     8192
#define T_STEPS 128
#define N_GROUPS 96     // 8*12
#define M_CHUNK 1536    // rows per pipeline chunk (8 chunks)

typedef __attribute__((ext_vector_type(16))) __bf16        v16bf;
typedef __attribute__((ext_vector_type(8)))  float         v8f;
typedef __attribute__((ext_vector_type(4)))  unsigned int  v4u;

union Frag {
  v16bf bf;
  v4u   u4[2];
};

__device__ __forceinline__ unsigned f32_to_bf16u(float f) {
  unsigned u = __float_as_uint(f);
  return (u + 0x7FFFu + ((u >> 16) & 1u)) >> 16;   // round-to-nearest-even
}
__device__ __forceinline__ unsigned pack2_bf16(float lo, float hi) {
  return (f32_to_bf16u(lo) & 0xFFFFu) | (f32_to_bf16u(hi) << 16);
}
__device__ __forceinline__ float bf16_lo(unsigned u) { return __uint_as_float(u << 16); }
__device__ __forceinline__ float bf16_hi(unsigned u) { return __uint_as_float(u & 0xFFFF0000u); }

// ---------------------------------------------------------------------------
// Pack fp32 row-major (M_TOTAL x K) -> A-fragment-swizzled bf16.
// 16-bit A 16x32 (ISA 7.12.2): lane l holds row m=(l&15), k-half (l>>4);
// VGPR v holds k pair k0 = (v&3)*2 + ((l>=16)?8:0) + ((v>=4)?16:0).
// One tile = 32 lanes * 8 dwords; dword index d -> fully coalesced writes.
// ---------------------------------------------------------------------------
__global__ __launch_bounds__(256)
void pack_a(const float* __restrict__ x, unsigned* __restrict__ dst, int K) {
  int d = blockIdx.x * 256 + threadIdx.x;
  int total = (M_TOTAL * K) >> 1;
  if (d >= total) return;
  int tile = d >> 8, r = d & 255, lane = r >> 3, v = r & 7;
  int KT = K >> 5;
  int mtile = tile / KT, kt = tile - mtile * KT;
  int m = (mtile << 4) + (lane & 15);
  int k = (kt << 5) + ((v & 3) << 1) + ((lane >> 4) << 3) + ((v >> 2) << 4);
  const float* p = x + (size_t)m * K + k;
  dst[d] = pack2_bf16(p[0], p[1]);
}

// ---------------------------------------------------------------------------
// Pack fp32 row-major weight (K x N) -> B-fragment-swizzled bf16.
// 16-bit B 32x16: lane l holds column n=(l&15); same k order as A:
// VGPR v holds k pair k0 = (v&3)*2 + ((l>=16)?8:0) + ((v>=4)?16:0).
// ---------------------------------------------------------------------------
__global__ __launch_bounds__(256)
void pack_b(const float* __restrict__ w, unsigned* __restrict__ dst, int K, int N) {
  int d = blockIdx.x * 256 + threadIdx.x;
  int total = (K * N) >> 1;
  if (d >= total) return;
  int tile = d >> 8, r = d & 255, lane = r >> 3, v = r & 7;
  int KT = K >> 5;
  int ntile = tile / KT, kt = tile - ntile * KT;
  int n = (ntile << 4) + (lane & 15);
  int k = (kt << 5) + ((v & 3) << 1) + ((lane >> 4) << 3) + ((v >> 2) << 4);
  float lo = w[(size_t)k * N + n];
  float hi = w[(size_t)(k + 1) * N + n];
  dst[d] = pack2_bf16(lo, hi);
}

// ---------------------------------------------------------------------------
// C(M x N) = A(M x K) * B(K x N) + bias, bf16 inputs (swizzled), f32 accum.
// Block = 256 threads = 8 waves stacked in M; wave tile = 32(M) x 64(N):
// 2 A-frags (global b128), 4 B-frags (LDS b128), 8 WMMAs per 32-wide K step.
// B staged via double-buffered LDS, 2 K-steps per stage -> 1 barrier / 16 WMMA.
// OUT_SWZ=0: fp32 row-major out. OUT_SWZ=1: bf16 out written directly into
// the A-swizzled layout (used for h so LN+GEMM2 need no re-layout).
// ---------------------------------------------------------------------------
template<int OUT_SWZ>
__global__ __launch_bounds__(256)
void gemm_bf16_wmma(const unsigned* __restrict__ A, const unsigned* __restrict__ B,
                    const float* __restrict__ bias, void* __restrict__ out,
                    int N, int K) {
  __shared__ unsigned ldsB[2][2048];     // 2 buffers x (2 ksteps x 4 ntiles x 256 dw)

  const int tid   = threadIdx.x;
  const int lane  = tid & 31;
  const int wave  = tid >> 5;
  const int KT    = K >> 5;              // 4..256, always even
  const int mWave = blockIdx.x * 256 + wave * 32;
  const int nWave = blockIdx.y * 64;
  const int mt0   = mWave >> 4;
  const int nt0   = nWave >> 4;

  // B staging: thread tid loads dwords (tid&63)*4..+3 of ntile j=tid>>6,
  // for each of the 2 staged k-steps (one global_load_b128 each).
  const int bj = tid >> 6;
  const int bo = (tid & 63) * 4;
  const unsigned* Bsrc = B + (((size_t)(nt0 + bj) * KT) << 8) + bo;

  const unsigned* Ap0 = A + (((size_t)(mt0    ) * KT) << 8) + lane * 8;
  const unsigned* Ap1 = A + (((size_t)(mt0 + 1) * KT) << 8) + lane * 8;

  v8f acc[2][4] = {};

  // Prologue: stage k-steps {0,1} into buffer 0.
  *(v4u*)&ldsB[0][bj * 256 + bo]        = *(const v4u*)(Bsrc);
  *(v4u*)&ldsB[0][1024 + bj * 256 + bo] = *(const v4u*)(Bsrc + 256);
  __syncthreads();

  for (int kt = 0; kt < KT; kt += 2) {
    const int cur = (kt >> 1) & 1;
    if (kt + 2 < KT) {
      // Stage next 2 k-steps into the other buffer (readers of it finished
      // before the barrier that ended the previous iteration).
      const size_t soff = (size_t)(kt + 2) << 8;
      *(v4u*)&ldsB[1 - cur][bj * 256 + bo]        = *(const v4u*)(Bsrc + soff);
      *(v4u*)&ldsB[1 - cur][1024 + bj * 256 + bo] = *(const v4u*)(Bsrc + soff + 256);
      if (kt + 4 < KT) {   // gfx1250 global_prefetch of the stage after that
        __builtin_prefetch(Bsrc + soff + 512, 0, 1);
        __builtin_prefetch(Ap0 + soff, 0, 1);
        __builtin_prefetch(Ap1 + soff, 0, 1);
      }
    }
#pragma unroll
    for (int u = 0; u < 2; ++u) {
      const size_t off = (size_t)(kt + u) << 8;
      Frag a0, a1, b0, b1, b2, b3;
      a0.u4[0] = *(const v4u*)(Ap0 + off); a0.u4[1] = *(const v4u*)(Ap0 + off + 4);
      a1.u4[0] = *(const v4u*)(Ap1 + off); a1.u4[1] = *(const v4u*)(Ap1 + off + 4);
      const unsigned* lb = &ldsB[cur][u * 1024 + lane * 8];
      b0.u4[0] = *(const v4u*)(lb +   0); b0.u4[1] = *(const v4u*)(lb +   4);
      b1.u4[0] = *(const v4u*)(lb + 256); b1.u4[1] = *(const v4u*)(lb + 260);
      b2.u4[0] = *(const v4u*)(lb + 512); b2.u4[1] = *(const v4u*)(lb + 516);
      b3.u4[0] = *(const v4u*)(lb + 768); b3.u4[1] = *(const v4u*)(lb + 772);

      acc[0][0] = __builtin_amdgcn_wmma_f32_16x16x32_bf16(false, a0.bf, false, b0.bf, (short)0, acc[0][0], false, false);
      acc[0][1] = __builtin_amdgcn_wmma_f32_16x16x32_bf16(false, a0.bf, false, b1.bf, (short)0, acc[0][1], false, false);
      acc[0][2] = __builtin_amdgcn_wmma_f32_16x16x32_bf16(false, a0.bf, false, b2.bf, (short)0, acc[0][2], false, false);
      acc[0][3] = __builtin_amdgcn_wmma_f32_16x16x32_bf16(false, a0.bf, false, b3.bf, (short)0, acc[0][3], false, false);
      acc[1][0] = __builtin_amdgcn_wmma_f32_16x16x32_bf16(false, a1.bf, false, b0.bf, (short)0, acc[1][0], false, false);
      acc[1][1] = __builtin_amdgcn_wmma_f32_16x16x32_bf16(false, a1.bf, false, b1.bf, (short)0, acc[1][1], false, false);
      acc[1][2] = __builtin_amdgcn_wmma_f32_16x16x32_bf16(false, a1.bf, false, b2.bf, (short)0, acc[1][2], false, false);
      acc[1][3] = __builtin_amdgcn_wmma_f32_16x16x32_bf16(false, a1.bf, false, b3.bf, (short)0, acc[1][3], false, false);
    }
    __syncthreads();
  }

  // Epilogue. C layout: VGPR r, lanes 0-15 -> M=r, lanes 16-31 -> M=8+r; N=lane&15.
  const int nLocal = lane & 15;
  const int mHalf  = (lane >> 4) << 3;
#pragma unroll
  for (int mf = 0; mf < 2; ++mf) {
#pragma unroll
    for (int nf = 0; nf < 4; ++nf) {
      const int n = nWave + nf * 16 + nLocal;
      const float bv = bias[n];
#pragma unroll
      for (int r = 0; r < 8; ++r) {
        const int m = mWave + mf * 16 + mHalf + r;
        const float val = acc[mf][nf][r] + bv;
        if (OUT_SWZ == 0) {
          ((float*)out)[(size_t)m * N + n] = val;
        } else {
          // bf16 element (m, k=n) into A-swizzled layout of width N.
          const int K2T = N >> 5;
          const int kt2 = n >> 5, kk = n & 31;
          const int l2  = (m & 15) + (((kk >> 3) & 1) << 4);
          const int v2  = (((kk >> 4) & 1) << 2) + ((kk >> 1) & 3);
          const size_t dw = (((size_t)((m >> 4) * K2T + kt2) << 5) + l2) * 8 + v2;
          ((unsigned short*)out)[dw * 2 + (kk & 1)] = (unsigned short)f32_to_bf16u(val);
        }
      }
    }
  }
}

// ---------------------------------------------------------------------------
// In-place LayerNorm(8192) + ReLU on the A-swizzled bf16 h buffer.
// One block (256 threads) per row; thread t owns k-tile t (32 elems, 16 dwords).
// ---------------------------------------------------------------------------
__global__ __launch_bounds__(256)
void ln_relu_swz(unsigned* __restrict__ h, const float* __restrict__ g,
                 const float* __restrict__ bb) {
  const int m = blockIdx.x;
  const int t = threadIdx.x;                 // 0..255 = ktile
  const int lane = t & 31, wave = t >> 5;
  const int l0 = m & 15;
  const size_t tileBase = (((size_t)(m >> 4) * 256) + t) << 8;
  unsigned* p0 = h + tileBase + l0 * 8;            // k-half 0
  unsigned* p1 = h + tileBase + (l0 + 16) * 8;     // k-half 1

  union { v4u v[4]; unsigned u[16]; } d;
  d.v[0] = *(const v4u*)p0;       d.v[1] = *(const v4u*)(p0 + 4);
  d.v[2] = *(const v4u*)p1;       d.v[3] = *(const v4u*)(p1 + 4);

  float s = 0.f, ss = 0.f;
#pragma unroll
  for (int i = 0; i < 16; ++i) {
    float lo = bf16_lo(d.u[i]), hi = bf16_hi(d.u[i]);
    s += lo + hi; ss += lo * lo + hi * hi;
  }
#pragma unroll
  for (int o = 16; o > 0; o >>= 1) {
    s  += __shfl_xor(s,  o, 32);
    ss += __shfl_xor(ss, o, 32);
  }
  __shared__ float rs[8], rss[8];
  if (lane == 0) { rs[wave] = s; rss[wave] = ss; }
  __syncthreads();
  float S = 0.f, SS = 0.f;
#pragma unroll
  for (int w = 0; w < 8; ++w) { S += rs[w]; SS += rss[w]; }
  const float mu   = S * (1.0f / 8192.0f);
  const float rstd = rsqrtf(SS * (1.0f / 8192.0f) - mu * mu + 1e-5f);

#pragma unroll
  for (int i = 0; i < 16; ++i) {
    const int v = i & 7, half = i >> 3;
    const int k0 = ((v & 3) << 1) + (half << 3) + ((v >> 2) << 4);
    const int gk = (t << 5) + k0;
    float lo = (bf16_lo(d.u[i]) - mu) * rstd * g[gk]     + bb[gk];
    float hi = (bf16_hi(d.u[i]) - mu) * rstd * g[gk + 1] + bb[gk + 1];
    lo = fmaxf(lo, 0.f); hi = fmaxf(hi, 0.f);
    d.u[i] = pack2_bf16(lo, hi);
  }
  *(v4u*)p0       = d.v[0]; *(v4u*)(p0 + 4) = d.v[1];
  *(v4u*)p1       = d.v[2]; *(v4u*)(p1 + 4) = d.v[3];
}

// ---------------------------------------------------------------------------
// Running mean over t (per (B,n) group of 128 rows) + concat [cummean, y].
// ---------------------------------------------------------------------------
__global__ __launch_bounds__(256)
void cumsum_cat(const float* __restrict__ y, float* __restrict__ out, int L) {
  const int c = blockIdx.x * 256 + threadIdx.x;
  const int grp = blockIdx.y;
  if (c >= L) return;
  float acc = 0.f;
  for (int tt = 0; tt < T_STEPS; ++tt) {
    const size_t row = (size_t)grp * T_STEPS + tt;
    const float v = y[row * L + c];
    acc += v;
    out[row * (2 * L) + c]     = acc / (float)(tt + 1);
    out[row * (2 * L) + L + c] = v;
  }
}

// ---------------------------------------------------------------------------
extern "C" void kernel_launch(void* const* d_in, const int* in_sizes, int n_in,
                              void* d_out, int out_size, void* d_ws, size_t ws_size,
                              hipStream_t stream) {
  (void)in_sizes; (void)n_in; (void)out_size; (void)ws_size;
  char* ws = (char*)d_ws;
  auto alignup = [](size_t x) { return (x + 255) & ~(size_t)255; };

  size_t o = 0;
  unsigned* xswz  = (unsigned*)(ws + o); o += alignup((size_t)M_TOTAL * 512 * 2); // 12.6 MB
  unsigned* w1swz = (unsigned*)(ws + o); o += alignup((size_t)512 * HID * 2);     //  8.4 MB
  unsigned* w2swz = (unsigned*)(ws + o); o += alignup((size_t)HID * 512 * 2);     //  8.4 MB
  unsigned* hswz  = (unsigned*)(ws + o); o += alignup((size_t)M_CHUNK * HID * 2); // 25.2 MB
  float*    ybuf  = (float*)   (ws + o); o += alignup((size_t)M_TOTAL * 512 * 4); // 25.2 MB
  float*    xbuf1 = (float*)   (ws + o); o += alignup((size_t)M_TOTAL * 256 * 4); // 12.6 MB
  float*    xbuf2 = (float*)   (ws + o);                                          // 25.2 MB

  const float* x = (const float*)d_in[0];
  int L = 128;
  for (int layer = 0; layer < 3; ++layer) {
    const float* w1 = (const float*)d_in[1 + 6 * layer + 0];
    const float* b1 = (const float*)d_in[1 + 6 * layer + 1];
    const float* g  = (const float*)d_in[1 + 6 * layer + 2];
    const float* bb = (const float*)d_in[1 + 6 * layer + 3];
    const float* w2 = (const float*)d_in[1 + 6 * layer + 4];
    const float* b2 = (const float*)d_in[1 + 6 * layer + 5];

    { int total = (M_TOTAL * L) >> 1;
      pack_a<<<(total + 255) / 256, 256, 0, stream>>>(x, xswz, L); }
    { int total = (L * HID) >> 1;
      pack_b<<<(total + 255) / 256, 256, 0, stream>>>(w1, w1swz, L, HID); }
    { int total = (HID * L) >> 1;
      pack_b<<<(total + 255) / 256, 256, 0, stream>>>(w2, w2swz, HID, L); }

    const int KT1 = L >> 5;
    for (int c = 0; c < M_TOTAL / M_CHUNK; ++c) {
      const int mOff = c * M_CHUNK;
      const unsigned* Achunk = xswz + (((size_t)(mOff >> 4) * KT1) << 8);
      gemm_bf16_wmma<1><<<dim3(M_CHUNK / 256, HID / 64), 256, 0, stream>>>(
          Achunk, w1swz, b1, hswz, HID, L);
      ln_relu_swz<<<M_CHUNK, 256, 0, stream>>>(hswz, g, bb);
      gemm_bf16_wmma<0><<<dim3(M_CHUNK / 256, L / 64), 256, 0, stream>>>(
          hswz, w2swz, b2, ybuf + (size_t)mOff * L, L, HID);
    }

    float* outp = (layer == 2) ? (float*)d_out : (layer == 0 ? xbuf1 : xbuf2);
    cumsum_cat<<<dim3((L + 255) / 256, N_GROUPS), 256, 0, stream>>>(ybuf, outp, L);
    x = outp;
    L <<= 1;
  }
}
